// TriFeaPred_OrigValid_43233140801876
// MI455X (gfx1250) — matrix operand order repaired and checked by reading, not compiled, not run on hardware
//
#include <hip/hip_runtime.h>

typedef __attribute__((ext_vector_type(16))) _Float16 v16h;
typedef __attribute__((ext_vector_type(8)))  _Float16 v8h;
typedef __attribute__((ext_vector_type(8)))  float    v8f;

namespace cfg {
constexpr int B = 4, N = 2048, NMETA = 10;
constexpr int NC1 = 1843, NK1 = 50;     // int(2048*0.9)
constexpr int NC2 = 1658, NK2 = 100;    // int(2048*0.81)
constexpr int BN  = B * N;              // 8192
constexpr int BN1 = B * NC1;            // 7372
constexpr int BN2 = B * NC2;            // 6632
constexpr long long M1 = (long long)BN1 * NK1;   // 368600
constexpr long long M2 = (long long)BN2 * NK2;   // 663200
// padded row counts (multiples of 128)
constexpr int MP1  = 368640;
constexpr int MP2  = 663296;
constexpr int MPB  = 8192;
constexpr int MPB1 = 7424;
constexpr int MPB2 = 6656;
constexpr int KNN_KMAX = 100;
constexpr int STAT_CHUNK = 1024;
}

// ---------------- xyz [B,N,3] -> point-major f16 feat [m][c], ld=4 ----------------
__global__ void k_xyz_to_feat(const float* __restrict__ xyz, _Float16* __restrict__ feat) {
    int m = blockIdx.x * 256 + threadIdx.x;
    if (m >= cfg::BN) return;
    feat[4 * m + 0] = (_Float16)xyz[3 * m + 0];
    feat[4 * m + 1] = (_Float16)xyz[3 * m + 1];
    feat[4 * m + 2] = (_Float16)xyz[3 * m + 2];
    feat[4 * m + 3] = (_Float16)0.f;
}

// ---------------- kNN (excluding self), sorted ascending ----------------
__global__ void k_knn(const float* __restrict__ coords, int Np, int K, int* __restrict__ out) {
    __shared__ float sd[64 * cfg::KNN_KMAX];
    __shared__ int   si[64 * cfg::KNN_KMAX];
    int t = threadIdx.x;
    int gid = blockIdx.x * 64 + t;
    if (gid >= cfg::B * Np) return;
    int b = gid / Np, i = gid - b * Np;
    const float* base = coords + (size_t)b * Np * 3;
    float xi = base[3 * i + 0], yi = base[3 * i + 1], zi = base[3 * i + 2];
    float* dl = sd + t * cfg::KNN_KMAX;
    int*   il = si + t * cfg::KNN_KMAX;
    int cnt = 0;
    for (int j = 0; j < Np; ++j) {
        if (j == i) continue;
        float dx = base[3 * j + 0] - xi, dy = base[3 * j + 1] - yi, dz = base[3 * j + 2] - zi;
        float d = dx * dx + dy * dy + dz * dz;
        if (cnt < K) {
            int p = cnt++;
            while (p > 0 && dl[p - 1] > d) { dl[p] = dl[p - 1]; il[p] = il[p - 1]; --p; }
            dl[p] = d; il[p] = j;
        } else if (d < dl[K - 1]) {
            int p = K - 1;
            while (p > 0 && dl[p - 1] > d) { dl[p] = dl[p - 1]; il[p] = il[p - 1]; --p; }
            dl[p] = d; il[p] = j;
        }
    }
    int* o = out + (size_t)gid * K;
    for (int q = 0; q < K; ++q) o[q] = il[q];
}

// ---------------- farthest point sampling: one block per batch ----------------
__global__ void k_fps(const float* __restrict__ coords, int Np, int S, int* __restrict__ fpsidx) {
    __shared__ float sv[256];
    __shared__ int   sj[256];
    int b = blockIdx.x, t = threadIdx.x;
    const float* base = coords + (size_t)b * Np * 3;
    float dloc[8];
#pragma unroll
    for (int q = 0; q < 8; ++q) dloc[q] = 1e10f;
    int far = 0;
    for (int s = 0; s < S; ++s) {
        if (t == 0) fpsidx[b * S + s] = far;
        float cx = base[3 * far + 0], cy = base[3 * far + 1], cz = base[3 * far + 2];
        float best = -1.0f; int bj = 0x7fffffff;
#pragma unroll
        for (int q = 0; q < 8; ++q) {
            int j = q * 256 + t;
            if (j < Np) {
                float dx = base[3 * j + 0] - cx, dy = base[3 * j + 1] - cy, dz = base[3 * j + 2] - cz;
                float d = dx * dx + dy * dy + dz * dz;
                float nd = fminf(dloc[q], d);
                dloc[q] = nd;
                if (nd > best || (nd == best && j < bj)) { best = nd; bj = j; }
            }
        }
        sv[t] = best; sj[t] = bj;
        __syncthreads();
        for (int off = 128; off > 0; off >>= 1) {
            if (t < off) {
                if (sv[t + off] > sv[t] || (sv[t + off] == sv[t] && sj[t + off] < sj[t])) {
                    sv[t] = sv[t + off]; sj[t] = sj[t + off];
                }
            }
            __syncthreads();
        }
        far = sj[0];
        __syncthreads();
    }
}

__global__ void k_gather_xyz(const float* __restrict__ coords, const int* __restrict__ fpsidx,
                             int Np, int S, float* __restrict__ newxyz) {
    int g = blockIdx.x * 256 + threadIdx.x;
    if (g >= cfg::B * S) return;
    int b = g / S;
    int j = fpsidx[g];
    const float* p = coords + ((size_t)b * Np + j) * 3;
    newxyz[3 * g + 0] = p[0];
    newxyz[3 * g + 1] = p[1];
    newxyz[3 * g + 2] = p[2];
}

__global__ void k_gather_idx(const int* __restrict__ knn, const int* __restrict__ fpsidx,
                             int Np, int S, int K, int* __restrict__ out) {
    long long g = (long long)blockIdx.x * 256 + threadIdx.x;
    long long total = (long long)cfg::B * S * K;
    if (g >= total) return;
    int kk = (int)(g % K);
    long long bs = g / K;
    int b = (int)(bs / S);
    int j = fpsidx[bs];
    out[g] = knn[((size_t)b * Np + j) * K + kk];
}

// ---------------- build grouped matrix (point-major f16): XG[m][0..2]=dxyz, [3..3+D)=feat ----------------
__global__ void k_group(const float* __restrict__ coords, const float* __restrict__ newxyz,
                        const int* __restrict__ idx, const _Float16* __restrict__ feat,
                        int Np, int S, int K, int D, int ldF,
                        _Float16* __restrict__ XG, int ldX) {
    long long g = (long long)blockIdx.x * 256 + threadIdx.x;
    long long Mt = (long long)cfg::B * S * K;
    if (g >= Mt) return;
    long long bs = g / K;
    int b = (int)(bs / S);
    int j = idx[g];
    const float* cj = coords + ((size_t)b * Np + j) * 3;
    const float* nc = newxyz + (size_t)bs * 3;
    _Float16* xrow = XG + (size_t)g * ldX;
    xrow[0] = (_Float16)(cj[0] - nc[0]);
    xrow[1] = (_Float16)(cj[1] - nc[1]);
    xrow[2] = (_Float16)(cj[2] - nc[2]);
    const _Float16* frow = feat + ((size_t)b * Np + j) * ldF;
    for (int c = 0; c < D; ++c) xrow[3 + c] = frow[c];
}

// ---------------- zero f16 channel columns [c0,c1) over Mp rows ----------------
__global__ void k_zero_cols_h(_Float16* __restrict__ X, int ldX, int c0, int c1, long long Mp) {
    long long g = (long long)blockIdx.x * 256 + threadIdx.x;
    int w = c1 - c0;
    long long total = Mp * w;
    if (g >= total) return;
    long long m = g / w;
    int c = c0 + (int)(g - m * w);
    X[m * (size_t)ldX + c] = (_Float16)0.f;
}

// ---------------- pad+convert weights: W[O,C] f32 -> WP[Op,Cp] f16 (zero-filled) ----------------
__global__ void k_padw(const float* __restrict__ W, int O, int C,
                       _Float16* __restrict__ WP, int Op, int Cp) {
    int g = blockIdx.x * 256 + threadIdx.x;
    if (g >= Op * Cp) return;
    int r = g / Cp, c = g - r * Cp;
    float v = (r < O && c < C) ? W[(size_t)r * C + c] : 0.f;
    WP[g] = (_Float16)v;
}

__global__ void k_padb(const float* __restrict__ Bi, int O, int hasB,
                       float* __restrict__ BP, int Op) {
    int g = blockIdx.x * 256 + threadIdx.x;
    if (g >= Op) return;
    BP[g] = (hasB && g < O) ? Bi[g] : 0.f;
}

// ---------------- WMMA GEMM (all-f16 inputs): Y[m][n] = sum_c X[m][c]*W[n][c] + BP[n] ----------------
// A = X tile (M=points, K=channels contiguous), B = W rows (N=out ch, K contiguous).
// Wave: 2 M-strips x 4 N-tiles; block 8 waves = 128 points x 128 out-channels.
__global__ __launch_bounds__(256) void k_gemm_wmma(
        const _Float16* __restrict__ XP, const _Float16* __restrict__ WP,
        const float* __restrict__ BP, float* __restrict__ Y,
        int Cp, int ldX, int ldY, int Nstore) {
    int lane = threadIdx.x & 31;
    int wave = threadIdx.x >> 5;
    int wrow = wave & 3;        // 4 M groups of 32 points
    int wcol = wave >> 2;       // 2 N groups of 64 channels
    int half = lane >> 4;
    int l16  = lane & 15;
    int mBase = blockIdx.x * 128 + wrow * 32;
    int nBase = blockIdx.y * 128 + wcol * 64;

    const _Float16* xa0 = XP + (size_t)(mBase + l16) * ldX + half * 16;
    const _Float16* xa1 = XP + (size_t)(mBase + 16 + l16) * ldX + half * 16;
    const _Float16* wb0 = WP + (size_t)(nBase +  0 + l16) * Cp + half * 16;
    const _Float16* wb1 = WP + (size_t)(nBase + 16 + l16) * Cp + half * 16;
    const _Float16* wb2 = WP + (size_t)(nBase + 32 + l16) * Cp + half * 16;
    const _Float16* wb3 = WP + (size_t)(nBase + 48 + l16) * Cp + half * 16;

    v8f zero = {0.f, 0.f, 0.f, 0.f, 0.f, 0.f, 0.f, 0.f};
    v8f acc[2][4];
#pragma unroll
    for (int s = 0; s < 2; ++s)
#pragma unroll
        for (int t = 0; t < 4; ++t) acc[s][t] = zero;

    for (int k0 = 0; k0 < Cp; k0 += 32) {
        v16h a[2];
#pragma unroll
        for (int s = 0; s < 2; ++s) {
            const _Float16* xs = (s == 0) ? xa0 : xa1;
            v8h r0 = *(const v8h*)(xs + k0);        // K run k0+half*16 + 0..7
            v8h r1 = *(const v8h*)(xs + k0 + 8);    // K run k0+half*16 + 8..15
#pragma unroll
            for (int j = 0; j < 8; ++j) { a[s][j] = r0[j]; a[s][8 + j] = r1[j]; }
        }
#pragma unroll
        for (int t = 0; t < 4; ++t) {
            const _Float16* wp = (t == 0) ? wb0 : (t == 1) ? wb1 : (t == 2) ? wb2 : wb3;
            v8h b0 = *(const v8h*)(wp + k0);
            v8h b1 = *(const v8h*)(wp + k0 + 8);
            v16h bf;
#pragma unroll
            for (int j = 0; j < 8; ++j) { bf[j] = b0[j]; bf[8 + j] = b1[j]; }
            acc[0][t] = __builtin_amdgcn_wmma_f32_16x16x32_f16(
                false, a[0], false, bf, (short)0, acc[0][t], false, false);
            acc[1][t] = __builtin_amdgcn_wmma_f32_16x16x32_f16(
                false, a[1], false, bf, (short)0, acc[1][t], false, false);
        }
    }
#pragma unroll
    for (int t = 0; t < 4; ++t) {
        int n0 = nBase + t * 16;
        if (n0 < Nstore) {               // uniform branch per wave
            int n = n0 + l16;
            float bias = BP[n];
#pragma unroll
            for (int s = 0; s < 2; ++s) {
#pragma unroll
                for (int r = 0; r < 8; ++r) {
                    int m = mBase + s * 16 + half * 8 + r;
                    Y[(size_t)m * ldY + n] = acc[s][t][r] + bias;
                }
            }
        }
    }
}

// ---------------- BN stats stage 1: per-(row-chunk, channel) partial sums ----------------
__global__ void k_bnstats1(const float* __restrict__ Y, int ldY, int O, long long Mreal,
                           float* __restrict__ part) {
    int blk = blockIdx.x, t = threadIdx.x;
    long long m0 = (long long)blk * cfg::STAT_CHUNK;
    long long m1 = m0 + cfg::STAT_CHUNK; if (m1 > Mreal) m1 = Mreal;
    float s = 0.f, q = 0.f;
    if (t < O) {
        for (long long m = m0; m < m1; ++m) {
            float v = Y[(size_t)m * ldY + t];
            s += v; q += v * v;
        }
    }
    part[(size_t)blk * 512 + t] = s;
    part[(size_t)blk * 512 + 256 + t] = q;
}

__global__ void k_bnstats2(const float* __restrict__ part, int nblk, long long Mreal,
                           float* __restrict__ stats) {
    __shared__ float ss[256], sq[256];
    int c = blockIdx.x, t = threadIdx.x;
    float s = 0.f, q = 0.f;
    for (int b = t; b < nblk; b += 256) {
        s += part[(size_t)b * 512 + c];
        q += part[(size_t)b * 512 + 256 + c];
    }
    ss[t] = s; sq[t] = q;
    __syncthreads();
    for (int off = 128; off > 0; off >>= 1) {
        if (t < off) { ss[t] += ss[t + off]; sq[t] += sq[t + off]; }
        __syncthreads();
    }
    if (t == 0) {
        float mean = ss[0] / (float)Mreal;
        float var  = sq[0] / (float)Mreal - mean * mean;
        if (var < 0.f) var = 0.f;
        stats[2 * c] = mean;
        stats[2 * c + 1] = rsqrtf(var + 1e-5f);
    }
}

// ---------------- BN normalize + ReLU: f32 Y -> f16 dst (zero-fills channel padding) ----------------
__global__ void k_bnrelu(const float* __restrict__ Y, const float* __restrict__ stats,
                         int ldY, int O, int Nstore, _Float16* __restrict__ dst, int ldD,
                         long long total) {
    long long g = (long long)blockIdx.x * 256 + threadIdx.x;
    if (g >= total) return;
    long long m = g / Nstore;
    int c = (int)(g - m * Nstore);
    float v = 0.f;
    if (c < O) {
        v = (Y[(size_t)m * ldY + c] - stats[2 * c]) * stats[2 * c + 1];
        v = v > 0.f ? v : 0.f;
    }
    dst[(size_t)m * ldD + c] = (_Float16)v;
}

// ---------------- f32 -> f16 convert (for non-BN conv2 output) ----------------
__global__ void k_cvt_h(const float* __restrict__ src, _Float16* __restrict__ dst, long long total) {
    long long g = (long long)blockIdx.x * 256 + threadIdx.x;
    if (g >= total) return;
    dst[g] = (_Float16)src[g];
}

// ---------------- max-pool over K neighbors (f16 point-major) ----------------
__global__ void k_maxpool(const _Float16* __restrict__ Y, int ldIn, int O, long long BS, int K,
                          _Float16* __restrict__ out, int ldOut) {
    long long g = (long long)blockIdx.x * 256 + threadIdx.x;
    long long total = BS * O;
    if (g >= total) return;
    int c = (int)(g % O);
    long long bs = g / O;
    const _Float16* p = Y + (size_t)bs * K * ldIn + c;
    float m = (float)p[0];
    for (int kk = 1; kk < K; ++kk) m = fmaxf(m, (float)p[(size_t)kk * ldIn]);
    out[(size_t)bs * ldOut + c] = (_Float16)m;
}

// ---------------- 3-NN + inverse-distance weights ----------------
__global__ void k_three_nn(const float* __restrict__ x1, const float* __restrict__ x2,
                           int N1, int N2, int* __restrict__ idx3, float* __restrict__ w3) {
    int g = blockIdx.x * 256 + threadIdx.x;
    if (g >= cfg::B * N1) return;
    int b = g / N1;
    float px = x1[3 * g + 0], py = x1[3 * g + 1], pz = x1[3 * g + 2];
    const float* base = x2 + (size_t)b * N2 * 3;
    float d0 = 3.4e38f, d1 = 3.4e38f, d2 = 3.4e38f;
    int i0 = 0, i1 = 0, i2 = 0;
    for (int j = 0; j < N2; ++j) {
        float dx = base[3 * j + 0] - px, dy = base[3 * j + 1] - py, dz = base[3 * j + 2] - pz;
        float d = dx * dx + dy * dy + dz * dz;
        if (d < d0)      { d2 = d1; i2 = i1; d1 = d0; i1 = i0; d0 = d; i0 = j; }
        else if (d < d1) { d2 = d1; i2 = i1; d1 = d;  i1 = j; }
        else if (d < d2) { d2 = d;  i2 = j; }
    }
    float r0 = 1.f / (d0 + 1e-8f), r1 = 1.f / (d1 + 1e-8f), r2 = 1.f / (d2 + 1e-8f);
    float inv = 1.f / (r0 + r1 + r2);
    idx3[3 * g + 0] = i0; idx3[3 * g + 1] = i1; idx3[3 * g + 2] = i2;
    w3[3 * g + 0] = r0 * inv; w3[3 * g + 1] = r1 * inv; w3[3 * g + 2] = r2 * inv;
}

// ---------------- interpolate f16 p2 into f16 X cols [off, off+D2) ----------------
__global__ void k_interp(const _Float16* __restrict__ p2, int ld2, const int* __restrict__ idx3,
                         const float* __restrict__ w3, int D2, int N1, int N2,
                         _Float16* __restrict__ X, int coff, int ldX) {
    long long g = (long long)blockIdx.x * 256 + threadIdx.x;
    long long total = (long long)cfg::B * N1 * D2;
    if (g >= total) return;
    long long m = g / D2;
    int c = (int)(g - m * D2);
    int b = (int)(m / N1);
    const int* ii = idx3 + 3 * m;
    const float* ww = w3 + 3 * m;
    const _Float16* pb = p2 + (size_t)b * N2 * ld2;
    float v = (float)pb[(size_t)ii[0] * ld2 + c] * ww[0]
            + (float)pb[(size_t)ii[1] * ld2 + c] * ww[1]
            + (float)pb[(size_t)ii[2] * ld2 + c] * ww[2];
    X[(size_t)m * ldX + coff + c] = (_Float16)v;
}

// ---------------- copy Cs f16 channels (point-major) ----------------
__global__ void k_copych(const _Float16* __restrict__ src, int ldS, _Float16* __restrict__ dst,
                         int ldD, int Cs, long long Mreal, int coff) {
    long long g = (long long)blockIdx.x * 256 + threadIdx.x;
    long long total = Mreal * Cs;
    if (g >= total) return;
    long long m = g / Cs;
    int c = (int)(g - m * Cs);
    dst[(size_t)m * ldD + coff + c] = src[(size_t)m * ldS + c];
}

// ---------------- emit: f32 Y (point-major [B*N][ldC]) -> out [B,O,N] ----------------
__global__ void k_emit(const float* __restrict__ Y, int ldC, int O, float* __restrict__ out) {
    long long g = (long long)blockIdx.x * 256 + threadIdx.x;
    long long total = (long long)cfg::B * O * cfg::N;
    if (g >= total) return;
    int b = (int)(g / ((long long)O * cfg::N));
    int r = (int)(g - (long long)b * O * cfg::N);
    int o = r / cfg::N, n = r - o * cfg::N;
    out[g] = Y[((size_t)b * cfg::N + n) * ldC + o];
}

extern "C" void kernel_launch(void* const* d_in, const int* in_sizes, int n_in,
                              void* d_out, int out_size, void* d_ws, size_t ws_size,
                              hipStream_t stream) {
    using namespace cfg;
    (void)in_sizes; (void)n_in; (void)out_size; (void)ws_size;

    size_t cur = 0;
    auto alloc = [&](size_t bytes) -> void* {
        void* p = (void*)((char*)d_ws + cur);
        cur = (cur + bytes + 255) & ~((size_t)255);
        return p;
    };
    _Float16*  FEAT0 = (_Float16*)alloc(sizeof(_Float16) * (size_t)MPB  * 4);
    _Float16*  FEAT1 = (_Float16*)alloc(sizeof(_Float16) * (size_t)MPB1 * 128);
    _Float16*  FEAT2 = (_Float16*)alloc(sizeof(_Float16) * (size_t)MPB2 * 256);
    _Float16*  L1NEW = (_Float16*)alloc(sizeof(_Float16) * (size_t)MPB1 * 128);
    _Float16*  L0NEW = (_Float16*)alloc(sizeof(_Float16) * (size_t)MPB  * 128);
    _Float16*  FC1H  = (_Float16*)alloc(sizeof(_Float16) * (size_t)MPB  * 128);
    _Float16*  EMBH  = (_Float16*)alloc(sizeof(_Float16) * (size_t)MPB  * 256);
    float*     XYZ1  = (float*)alloc(sizeof(float) * 3 * BN1);
    float*     XYZ2  = (float*)alloc(sizeof(float) * 3 * BN2);
    int*       FPSI  = (int*)  alloc(sizeof(int) * BN1);
    int*       KNNI  = (int*)  alloc(sizeof(int) * 737200);
    int*       GI    = (int*)  alloc(sizeof(int) * (size_t)M2);
    int*       NN3I  = (int*)  alloc(sizeof(int) * 3 * BN);
    float*     NN3W  = (float*)alloc(sizeof(float) * 3 * BN);
    float*     STATS = (float*)alloc(sizeof(float) * 512);
    float*     PART  = (float*)alloc(sizeof(float) * 1024 * 512);
    _Float16*  WPH   = (_Float16*)alloc(sizeof(_Float16) * 256 * 384);
    float*     BPH   = (float*)alloc(sizeof(float) * 256);
    _Float16*  XGH   = (_Float16*)alloc(sizeof(_Float16) * (size_t)MP2 * 160); // also 384xMPB1, 160xMPB
    float*     YA    = (float*)alloc(sizeof(float) * (size_t)MP2 * 256);       // f32 GEMM output arena
    _Float16*  HA    = (_Float16*)alloc(sizeof(_Float16) * (size_t)MP2 * 256); // f16 activation ping
    _Float16*  HB    = (_Float16*)alloc(sizeof(_Float16) * (size_t)MP2 * 256); // f16 activation pong

    auto F = [&](int i) { return (const float*)d_in[i]; };
    const float* xyz0 = F(0);

    // GEMM: X f16 point-major [Mp, ldX] x W[O,C] -> YA f32 [Mp, ldY=Nstore]
    auto gemm = [&](const _Float16* Xp, int ldX, const float* Wp, const float* bp,
                    float* Yp, int ldY, int O, int C, long long Mp) {
        int Op = ((O + 127) / 128) * 128;
        int Cp = ((C + 31) / 32) * 32;
        int Nstore = ((O + 31) / 32) * 32;
        k_padw<<<(unsigned)((Op * Cp + 255) / 256), 256, 0, stream>>>(Wp, O, C, WPH, Op, Cp);
        k_padb<<<1, 256, 0, stream>>>(bp, O, bp ? 1 : 0, BPH, Op);
        dim3 g((unsigned)(Mp / 128), (unsigned)(Op / 128));
        k_gemm_wmma<<<g, 256, 0, stream>>>(Xp, WPH, BPH, Yp, Cp, ldX, ldY, Nstore);
    };
    // BN(train) + ReLU: YA f32 -> dst f16 (with zero channel padding up to Nstore)
    auto bnrelu = [&](const float* Yp, int ldY, int O, long long Mreal,
                      _Float16* dst, int ldD) {
        int nblk = (int)((Mreal + STAT_CHUNK - 1) / STAT_CHUNK);
        int Nstore = ((O + 31) / 32) * 32;
        k_bnstats1<<<(unsigned)nblk, 256, 0, stream>>>(Yp, ldY, O, Mreal, PART);
        k_bnstats2<<<(unsigned)O, 256, 0, stream>>>(PART, nblk, Mreal, STATS);
        long long tot = Mreal * Nstore;
        k_bnrelu<<<(unsigned)((tot + 255) / 256), 256, 0, stream>>>(Yp, STATS, ldY, O, Nstore,
                                                                    dst, ldD, tot);
    };
    auto zero_cols = [&](_Float16* Xp, int ldX, int c0, int c1, long long Mp) {
        if (c1 <= c0) return;
        long long cnt = Mp * (c1 - c0);
        k_zero_cols_h<<<(unsigned)((cnt + 255) / 256), 256, 0, stream>>>(Xp, ldX, c0, c1, Mp);
    };

    k_xyz_to_feat<<<(BN + 255) / 256, 256, 0, stream>>>(xyz0, FEAT0);

    // ---- generic set-abstraction stage ----
    auto set_abs = [&](const float* coords, const _Float16* featIn, int ldF,
                       int Np, int S, int K, int D,
                       const float* w0, int o0, const float* w1, int o1, const float* w2, int o2,
                       float* xyzOut, _Float16* featOut, int ldOut, long long Mp) {
        int BNp = B * Np;
        long long Mt = (long long)B * S * K;
        int Cin = 3 + D, Cp = ((Cin + 31) / 32) * 32;
        int ldG = Cp;
        int ld1 = ((o0 + 31) / 32) * 32;
        int ld2 = ((o1 + 31) / 32) * 32;
        int ld3 = ((o2 + 31) / 32) * 32;
        k_knn<<<(unsigned)((BNp + 63) / 64), 64, 0, stream>>>(coords, Np, K, KNNI);
        k_fps<<<B, 256, 0, stream>>>(coords, Np, S, FPSI);
        k_gather_xyz<<<(unsigned)((B * S + 255) / 256), 256, 0, stream>>>(coords, FPSI, Np, S, xyzOut);
        k_gather_idx<<<(unsigned)((Mt + 255) / 256), 256, 0, stream>>>(KNNI, FPSI, Np, S, K, GI);
        k_group<<<(unsigned)((Mt + 255) / 256), 256, 0, stream>>>(coords, xyzOut, GI, featIn,
                                                                  Np, S, K, D, ldF, XGH, ldG);
        zero_cols(XGH, ldG, Cin, Cp, Mp);
        gemm(XGH, ldG, w0, nullptr, YA, ld1, o0, Cin, Mp); bnrelu(YA, ld1, o0, Mt, HA, ld1);
        gemm(HA, ld1, w1, nullptr, YA, ld2, o1, o0, Mp);   bnrelu(YA, ld2, o1, Mt, HB, ld2);
        gemm(HB, ld2, w2, nullptr, YA, ld3, o2, o1, Mp);   bnrelu(YA, ld3, o2, Mt, HA, ld3);
        long long pt = (long long)B * S * o2;
        k_maxpool<<<(unsigned)((pt + 255) / 256), 256, 0, stream>>>(HA, ld3, o2, (long long)B * S, K,
                                                                    featOut, ldOut);
    };

    // sa1: 2048 -> 1843 centers, k=50, D=3, 6->64->64->128
    set_abs(xyz0, FEAT0, 4, N, NC1, NK1, 3, F(1), 64, F(3), 64, F(5), 128, XYZ1, FEAT1, 128, MP1);
    // sa2: 1843 -> 1658 centers, k=100, D=128, 131->128->128->256
    set_abs(XYZ1, FEAT1, 128, NC1, NC2, NK2, 128, F(7), 128, F(9), 128, F(11), 256, XYZ2, FEAT2, 256, MP2);

    // ---- fp2: interp l2 (256ch @ NC2) onto NC1, concat with l1 (128ch); C=384 ----
    {
        int N1 = NC1, N2 = NC2;
        long long Mr = BN1, Mp = MPB1;
        k_three_nn<<<(unsigned)((B * N1 + 255) / 256), 256, 0, stream>>>(XYZ1, XYZ2, N1, N2, NN3I, NN3W);
        long long ct = Mr * 128;
        k_copych<<<(unsigned)((ct + 255) / 256), 256, 0, stream>>>(FEAT1, 128, XGH, 384, 128, Mr, 0);
        long long it = Mr * 256;
        k_interp<<<(unsigned)((it + 255) / 256), 256, 0, stream>>>(FEAT2, 256, NN3I, NN3W, 256, N1, N2,
                                                                   XGH, 128, 384);
        gemm(XGH, 384, F(13), nullptr, YA, 256, 256, 384, Mp);  bnrelu(YA, 256, 256, Mr, HA, 256);
        gemm(HA, 256, F(15), nullptr, YA, 256, 256, 256, Mp);   bnrelu(YA, 256, 256, Mr, HB, 256);
        gemm(HB, 256, F(17), nullptr, YA, 128, 128, 256, Mp);   bnrelu(YA, 128, 128, Mr, L1NEW, 128);
    }
    // ---- fp1: interp l1new (128ch @ NC1) onto N, concat with [xyz,xyz] (6ch); C=134 ----
    {
        int N1 = N, N2 = NC1;
        long long Mr = BN, Mp = MPB;
        k_three_nn<<<(unsigned)((B * N1 + 255) / 256), 256, 0, stream>>>(xyz0, XYZ1, N1, N2, NN3I, NN3W);
        long long c3 = Mr * 3;
        k_copych<<<(unsigned)((c3 + 255) / 256), 256, 0, stream>>>(FEAT0, 4, XGH, 160, 3, Mr, 0);
        k_copych<<<(unsigned)((c3 + 255) / 256), 256, 0, stream>>>(FEAT0, 4, XGH, 160, 3, Mr, 3);
        long long it = Mr * 128;
        k_interp<<<(unsigned)((it + 255) / 256), 256, 0, stream>>>(L1NEW, 128, NN3I, NN3W, 128, N1, N2,
                                                                   XGH, 6, 160);
        zero_cols(XGH, 160, 134, 160, Mp);
        gemm(XGH, 160, F(19), nullptr, YA, 128, 128, 134, Mp);  bnrelu(YA, 128, 128, Mr, HA, 128);
        gemm(HA, 128, F(21), nullptr, YA, 128, 128, 128, Mp);   bnrelu(YA, 128, 128, Mr, HB, 128);
        gemm(HB, 128, F(23), nullptr, YA, 128, 128, 128, Mp);   bnrelu(YA, 128, 128, Mr, L0NEW, 128);
    }

    // ---- trunk ----
    gemm(L0NEW, 128, F(25), nullptr, YA, 128, 128, 128, MPB); bnrelu(YA, 128, 128, BN, FC1H, 128);
    gemm(FC1H, 128, F(27), F(28), YA, 256, 256, 128, MPB);    // bias, no BN/relu
    k_cvt_h<<<(unsigned)(((long long)MPB * 256 + 255) / 256), 256, 0, stream>>>(YA, EMBH,
                                                                                (long long)MPB * 256);

    // ---- heads ----
    float* out = (float*)d_out;
    auto run_head = [&](const float* w0, const float* w1, const float* w2, const float* w3,
                        const float* b3, int c1, int c2, int c3, int c4, float* outPtr) {
        int ld3 = ((c3 + 31) / 32) * 32;
        int ld4 = ((c4 + 31) / 32) * 32;
        gemm(EMBH, 256, w0, nullptr, YA, 256, c1, 256, MPB); bnrelu(YA, 256, c1, BN, HA, 256);
        gemm(HA, 256, w1, nullptr, YA, 128, c2, c1, MPB);    bnrelu(YA, 128, c2, BN, HB, 128);
        gemm(HB, 128, w2, nullptr, YA, ld3, c3, c2, MPB);    bnrelu(YA, ld3, c3, BN, HA, ld3);
        gemm(HA, ld3, w3, b3, YA, ld4, c4, c3, MPB);         // final conv with bias
        long long tot = (long long)B * c4 * N;
        k_emit<<<(unsigned)((tot + 255) / 256), 256, 0, stream>>>(YA, ld4, c4, outPtr);
    };
    run_head(F(29), F(31), F(33), F(35), F(36), 256, 128, 32, 3,     out);
    run_head(F(37), F(39), F(41), F(43), F(44), 256, 128, 32, 2,     out + (size_t)B * 3 * N);
    run_head(F(45), F(47), F(49), F(51), F(52), 256, 128, 64, NMETA, out + (size_t)B * 5 * N);
}
